// ParallelTransformerBlock_6536940225033
// MI455X (gfx1250) — compile-verified, matrix operand order checked by reading, not run on hardware
//
#include <hip/hip_runtime.h>
#include <cstdint>

#define DIMC   2048
#define SEQ    2048
#define BATCH  2
#define NHEADS 16
#define DHEAD  64
#define ROWS   (BATCH*SEQ)          // 4096
#define FUSED  17536
#define QKVN   1152                 // q(1024) + k(64) + v(64)
#define FFI    8192
#define FF2    (2*FFI)              // 16384

typedef __attribute__((ext_vector_type(16))) __bf16 v16bf;
typedef __attribute__((ext_vector_type(8)))  float  v8f;

union Frag16 { uint4 u4[2]; v16bf v; };

__device__ __forceinline__ v8f wmma_bf16(const Frag16& a, const Frag16& b, v8f c) {
  // D = A(16x32 bf16) * B(32x16 bf16) + C(16x16 f32)
  return __builtin_amdgcn_wmma_f32_16x16x32_bf16(false, a.v, false, b.v,
                                                 (short)0, c, false, false);
}

// ---------------- Tensor Data Mover (CDNA5) ----------------
#ifndef __has_builtin
#define __has_builtin(x) 0
#endif
#if defined(__gfx1250__) && __has_builtin(__builtin_amdgcn_tensor_load_to_lds) && \
    __has_builtin(__builtin_amdgcn_s_wait_tensorcnt)
#define USE_TDM_DEV 1
#if __has_include(<hip/amd_detail/amd_gfx1250_TDM.h>)
#define TDM_ARITY6 1      // amdgpu-toolchain (clang-23 / therock-10.0): 6-arg builtin
#endif
#endif

#ifdef USE_TDM_DEV
typedef __attribute__((ext_vector_type(4))) unsigned int tdm_u32x4;
typedef __attribute__((ext_vector_type(8))) int          tdm_i32x8;
typedef __attribute__((ext_vector_type(4))) int          tdm_i32x4;

// 2-D tile DMA (2-byte elements): global row stride stride0 (elements) -> LDS,
// with LDS row padding via pad_enable: pad_amount=3 => insert 4 DWORDs (16B =
// 8 elements) after every pad_interval'th chunk, reproducing the padded LDS
// row strides used by the WMMA fragment readers.
__device__ __forceinline__ void tdm_load_tile_2d(unsigned int lds_byte_off,
                                                 const void* gsrc,
                                                 unsigned int tdim0, unsigned int tdim1,
                                                 unsigned int tile0, unsigned int tile1,
                                                 unsigned int stride0,
                                                 unsigned int pad_interval,
                                                 unsigned int pad_amount) {
  unsigned long long ga = (unsigned long long)(uintptr_t)gsrc;
  tdm_u32x4 g0;
  g0[0] = 1u;                                   // count=1 (valid user descriptor)
  g0[1] = lds_byte_off;                         // lds_addr (bytes, wave-relative)
  g0[2] = (unsigned int)ga;                     // global_addr[31:0]
  g0[3] = (unsigned int)((ga >> 32) & 0x01FFFFFFu) | (2u << 30); // addr[56:32] | type=2
  tdm_i32x8 g1;
  g1[0] = (int)((1u << 16)                      // data_size = 1 -> 2 bytes
              | (1u << 20)                      // pad_enable
              | ((pad_interval & 7u) << 22)
              | ((pad_amount & 127u) << 25));
  g1[1] = (int)((tdim0 & 0xFFFFu) << 16);       // atomic_barrier_addr=0 | tensor_dim0 lo
  g1[2] = (int)(((tdim0 >> 16) & 0xFFFFu) | ((tdim1 & 0xFFFFu) << 16));
  g1[3] = (int)(((tdim1 >> 16) & 0xFFFFu) | ((tile0 & 0xFFFFu) << 16));
  g1[4] = (int)(tile1 & 0xFFFFu);               // tile_dim1 | tile_dim2=0
  g1[5] = (int)stride0;                         // tensor_dim0_stride[31:0]
  g1[6] = 0;                                    // stride hi | tensor_dim1_stride lo
  g1[7] = 0;
  tdm_i32x4 gz = {0, 0, 0, 0};
#ifdef TDM_ARITY6
  tdm_i32x8 gz8 = {0, 0, 0, 0, 0, 0, 0, 0};
  __builtin_amdgcn_tensor_load_to_lds(g0, g1, gz, gz, gz8, 0);
#else
  __builtin_amdgcn_tensor_load_to_lds(g0, g1, gz, gz, 0);
#endif
}
#endif

// ---------------- fused cast + transpose: W[K][N] f32 -> W^T[N][K] bf16 -----
__global__ __launch_bounds__(256)
void cast_transpose_kernel(const float* __restrict__ in, __bf16* __restrict__ outT,
                           int K, int N) {
  __shared__ __bf16 st[64][72];
  int k0 = blockIdx.y * 64, n0 = blockIdx.x * 64;
  int tx = threadIdx.x & 15, ty = threadIdx.x >> 4;       // 16 x 16 threads
#pragma unroll
  for (int i = 0; i < 4; ++i) {
    int r = i * 16 + ty;                                  // k within tile
    int c = tx * 4;                                       // n within tile
    float4 f = *(const float4*)(in + (size_t)(k0 + r) * N + n0 + c);
    st[r][c + 0] = (__bf16)f.x; st[r][c + 1] = (__bf16)f.y;
    st[r][c + 2] = (__bf16)f.z; st[r][c + 3] = (__bf16)f.w;
  }
  __syncthreads();
#pragma unroll
  for (int i = 0; i < 4; ++i) {
    int n = i * 16 + ty;                                  // n within tile
    int k = tx * 4;                                       // k within tile
    union { unsigned long long u; __bf16 h[4]; } d;
    d.h[0] = st[k + 0][n]; d.h[1] = st[k + 1][n];
    d.h[2] = st[k + 2][n]; d.h[3] = st[k + 3][n];
    *(unsigned long long*)(outT + (size_t)(n0 + n) * K + k0 + k) = d.u;
  }
}

// ---------------- LayerNorm (beta = 0) -> bf16 ----------------
__global__ __launch_bounds__(256)
void ln_kernel(const float* __restrict__ x, const float* __restrict__ gamma,
               __bf16* __restrict__ xn) {
  __shared__ float red[256];
  int row = blockIdx.x;
  const float* xr = x + (size_t)row * DIMC;
  float s = 0.f, s2 = 0.f;
  for (int c = threadIdx.x; c < DIMC; c += 256) { float v = xr[c]; s += v; s2 += v * v; }
  red[threadIdx.x] = s; __syncthreads();
  for (int off = 128; off > 0; off >>= 1) {
    if ((int)threadIdx.x < off) red[threadIdx.x] += red[threadIdx.x + off];
    __syncthreads();
  }
  float mean = red[0] * (1.f / DIMC); __syncthreads();
  red[threadIdx.x] = s2; __syncthreads();
  for (int off = 128; off > 0; off >>= 1) {
    if ((int)threadIdx.x < off) red[threadIdx.x] += red[threadIdx.x + off];
    __syncthreads();
  }
  float var  = red[0] * (1.f / DIMC) - mean * mean;
  float rstd = rsqrtf(var + 1e-5f);
  for (int c = threadIdx.x; c < DIMC; c += 256)
    xn[(size_t)row * DIMC + c] = (__bf16)((xr[c] - mean) * rstd * gamma[c]);
}

// ---------------- bf16 GEMM, fp32 accumulate, double-buffered TDM tiles -----
// C[M,N] = A[M,K](row major, lda) * Bt[N,K](row major, ldbt)^T
// MODE 0: Cf = v   MODE 1: Cf += v   MODE 2: Cb = (bf16)v
#define BM 256
#define BN 128
#define BK 32
#define LSA 40   // padded LDS row stride (elements); 80B keeps 16B alignment

template <int MODE>
__global__ __launch_bounds__(256)
void gemm_bf16_kernel(const __bf16* __restrict__ A, int lda,
                      const __bf16* __restrict__ Bt, int ldbt,
                      int K,
                      float* __restrict__ Cf, __bf16* __restrict__ Cb,
                      int ldc) {
  __shared__ __bf16 sA[2][BM][LSA];    // A tile (M x K), padded rows  (40 KB)
  __shared__ __bf16 sB[2][BN][LSA];    // B tile (N x K), padded rows  (20 KB)
  int tid = threadIdx.x;
  int lane = tid & 31, w = tid >> 5;
  int wm = w >> 1, wn = w & 1;                    // 4 x 2 wave grid -> 64x64 per wave
  int hlf = lane >> 4, l16 = lane & 15, kb = hlf * 8;
  int m0 = blockIdx.y * BM, n0 = blockIdx.x * BN;

  const v8f VZ = {0.f,0.f,0.f,0.f,0.f,0.f,0.f,0.f};
  v8f c[4][4];
#pragma unroll
  for (int i = 0; i < 4; ++i)
#pragma unroll
    for (int j = 0; j < 4; ++j) c[i][j] = VZ;

  auto stage = [&](int buf, int k0) {
#ifdef USE_TDM_DEV
    // One Tensor-DMA per tile; A on wave 0, B on wave 1 (each waits its own cnt)
    if (w == 0)
      tdm_load_tile_2d((unsigned int)(uintptr_t)&sA[buf][0][0],
                       A + (size_t)m0 * lda + k0,
                       (unsigned int)lda, 1u << 20, BK, BM,
                       (unsigned int)lda, 3u, 3u);       // pad -> 40-elem stride
    else if (w == 1)
      tdm_load_tile_2d((unsigned int)(uintptr_t)&sB[buf][0][0],
                       Bt + (size_t)n0 * ldbt + k0,
                       (unsigned int)ldbt, 1u << 20, BK, BN,
                       (unsigned int)ldbt, 3u, 3u);
#else
    // fallback: 16B chunks; A = 1024 chunks (4/thread), B = 512 chunks (2/thread)
#pragma unroll
    for (int i = 0; i < 4; ++i) {
      int ch = tid + i * 256;
      int r = ch >> 2, cc = (ch & 3) * 8;
      *(uint4*)&sA[buf][r][cc] = *(const uint4*)(A + (size_t)(m0 + r) * lda + k0 + cc);
    }
#pragma unroll
    for (int i = 0; i < 2; ++i) {
      int ch = tid + i * 256;
      int r = ch >> 2, cc = (ch & 3) * 8;
      *(uint4*)&sB[buf][r][cc] = *(const uint4*)(Bt + (size_t)(n0 + r) * ldbt + k0 + cc);
    }
#endif
  };

  stage(0, 0);
#ifdef USE_TDM_DEV
  if (w <= 1) __builtin_amdgcn_s_wait_tensorcnt(0);
#endif
  int buf = 0;
  for (int k0 = 0; k0 < K; k0 += BK) {
    __syncthreads();                       // tile 'buf' visible, prev readers done
    if (k0 + BK < K) stage(buf ^ 1, k0 + BK);

    // fragment loads per the CDNA5 16-bit A layout:
    // lanes 0-15: row M=l16, K 0..7 & 16..23 ; lanes 16-31: K 8..15 & 24..31
    Frag16 af[4], bfm[4];
#pragma unroll
    for (int mt = 0; mt < 4; ++mt) {
      int r = wm * 64 + mt * 16 + l16;
      af[mt].u4[0] = *(const uint4*)&sA[buf][r][kb];
      af[mt].u4[1] = *(const uint4*)&sA[buf][r][kb + 16];
    }
#pragma unroll
    for (int nt = 0; nt < 4; ++nt) {
      int r = wn * 64 + nt * 16 + l16;
      bfm[nt].u4[0] = *(const uint4*)&sB[buf][r][kb];
      bfm[nt].u4[1] = *(const uint4*)&sB[buf][r][kb + 16];
    }
#pragma unroll
    for (int mt = 0; mt < 4; ++mt)
#pragma unroll
      for (int nt = 0; nt < 4; ++nt)
        c[mt][nt] = wmma_bf16(af[mt], bfm[nt], c[mt][nt]);

#ifdef USE_TDM_DEV
    if (w <= 1 && k0 + BK < K) __builtin_amdgcn_s_wait_tensorcnt(0);
#endif
    buf ^= 1;
  }

  // C layout: VGPR r, lanes 0-15 -> M=r, N=lane; lanes 16-31 -> M=r+8
#pragma unroll
  for (int mt = 0; mt < 4; ++mt)
#pragma unroll
    for (int nt = 0; nt < 4; ++nt)
#pragma unroll
      for (int r = 0; r < 8; ++r) {
        int m = m0 + wm * 64 + mt * 16 + r + 8 * hlf;
        int n = n0 + wn * 64 + nt * 16 + l16;
        size_t idx = (size_t)m * ldc + n;
        float v = c[mt][nt][r];
        if constexpr (MODE == 0)      Cf[idx] = v;
        else if constexpr (MODE == 1) Cf[idx] += v;
        else                          Cb[idx] = (__bf16)v;
      }
}

// ---------------- rope + scale + split -> bf16 q, k, v^T ----------------
__global__ __launch_bounds__(256)
void rope_kernel(const float* __restrict__ qkv,
                 __bf16* __restrict__ qb, __bf16* __restrict__ kbuf,
                 __bf16* __restrict__ vT) {
  int row = blockIdx.x;                 // b*SEQ + s
  int b   = row >> 11;
  int s   = row & (SEQ - 1);
  const float* pr = qkv + (size_t)row * QKVN;
  for (int c = threadIdx.x; c < QKVN; c += 256) {
    float val = pr[c];
    if (c < 1024) {                     // q: per-head dim d = c & 63
      int d = c & 63, dh2 = d & 31;
      float fr = __powf(10000.f, -(float)(2 * dh2) * (1.f / 64.f));
      float th = (float)s * fr;
      float sn = __sinf(th), cs = __cosf(th);
      float part = pr[(d < 32) ? c + 32 : c - 32];
      float rot  = (d < 32) ? -part : part;
      qb[(size_t)row * 1024 + c] = (__bf16)((val * cs + rot * sn) * 0.125f);
    } else if (c < 1088) {              // k
      int d = c - 1024, dh2 = d & 31;
      float fr = __powf(10000.f, -(float)(2 * dh2) * (1.f / 64.f));
      float th = (float)s * fr;
      float sn = __sinf(th), cs = __cosf(th);
      float part = pr[(d < 32) ? c + 32 : c - 32];
      float rot  = (d < 32) ? -part : part;
      kbuf[(size_t)row * 64 + d] = (__bf16)(val * cs + rot * sn);
    } else {                            // v, stored transposed: vT[b][d][s]
      int d = c - 1088;
      vT[((size_t)(b * 64 + d) << 11) + s] = (__bf16)val;
    }
  }
}

// ---------------- SwiGLU, in place over first half ----------------
__global__ __launch_bounds__(256)
void glu_kernel(__bf16* __restrict__ ffb) {
  size_t i = (size_t)blockIdx.x * 256 + threadIdx.x;   // over ROWS*FFI
  size_t r = i >> 13, cc = i & (FFI - 1);
  size_t base = r * (size_t)FF2;
  float xv = (float)ffb[base + cc];
  float g  = (float)ffb[base + FFI + cc];
  float act = g / (1.f + __expf(-g));                  // silu(g)
  ffb[base + cc] = (__bf16)(xv * act);
}

// ---------------- causal multi-query flash attention ----------------
// grid (SEQ/128, NHEADS, BATCH), 8 waves, each wave owns a 16-row query tile.
// K and V^T tiles staged by TDM (double buffered), one barrier per j-block.
// Row sums computed on the matrix pipe: rowsum(P) = P @ ones via one WMMA.
__global__ __launch_bounds__(256)
void attn_kernel(const __bf16* __restrict__ Q, const __bf16* __restrict__ Kc,
                 const __bf16* __restrict__ vT, const unsigned char* __restrict__ msk,
                 __bf16* __restrict__ O) {
  __shared__ __bf16 sK[2][32][72];       // 32 keys x 64 dims (padded)
  __shared__ __bf16 sVT[2][64][40];      // V^T: 64 dims x 32 keys (padded)
  __shared__ __bf16 sP[8][16][40];       // per-wave P scratch 16x32
  int tid = threadIdx.x, lane = tid & 31, w = tid >> 5;
  int hlf = lane >> 4, l16 = lane & 15, kb = hlf * 8;
  int m0b = blockIdx.x * 128;
  int h   = blockIdx.y;
  int b   = blockIdx.z;
  int m0w = m0b + w * 16;
  int jmax = m0b + 128;

  auto stage = [&](int buf, int j0) {
#ifdef USE_TDM_DEV
    if (w == 0)      // K tile: 32 keys x 64 elems (128B = 32 DW) -> interval 4
      tdm_load_tile_2d((unsigned int)(uintptr_t)&sK[buf][0][0],
                       Kc + (size_t)(b * SEQ + j0) * DHEAD,
                       1u << 20, 1u << 20, DHEAD, 32, DHEAD, 4u, 3u); // -> 72 stride
    else if (w == 1) // V^T tile: 64 dims x 32 keys (64B = 16 DW) -> interval 3
      tdm_load_tile_2d((unsigned int)(uintptr_t)&sVT[buf][0][0],
                       vT + ((size_t)b * 64 << 11) + j0,
                       1u << 20, 1u << 20, 32, 64, SEQ, 3u, 3u);      // -> 40 stride
#else
    {
      int key = tid >> 3, d0 = (tid & 7) * 8;
      *(uint4*)&sK[buf][key][d0] =
          *(const uint4*)(Kc + (size_t)(b * SEQ + j0 + key) * DHEAD + d0);
      int dd = tid >> 2, s0 = (tid & 3) * 8;
      *(uint4*)&sVT[buf][dd][s0] =
          *(const uint4*)(vT + (((size_t)b * 64 + dd) << 11) + j0 + s0);
    }
#endif
  };

  // q fragments (K-dim = 64 -> two K=32 fragments), loaded once
  Frag16 qf[2];
  {
    const __bf16* qp = Q + (size_t)(b * SEQ + m0w + l16) * (NHEADS * DHEAD) + h * DHEAD;
    qf[0].u4[0] = *(const uint4*)(qp + kb);
    qf[0].u4[1] = *(const uint4*)(qp + kb + 16);
    qf[1].u4[0] = *(const uint4*)(qp + 32 + kb);
    qf[1].u4[1] = *(const uint4*)(qp + 32 + kb + 16);
  }
  Frag16 ones;                           // B-matrix of 1.0 bf16 for row sums
  ones.u4[0] = uint4{0x3f803f80u, 0x3f803f80u, 0x3f803f80u, 0x3f803f80u};
  ones.u4[1] = ones.u4[0];

  const v8f VZ = {0.f,0.f,0.f,0.f,0.f,0.f,0.f,0.f};
  v8f o[4];
#pragma unroll
  for (int nt = 0; nt < 4; ++nt) o[nt] = VZ;
  float mrow[8], lrow[8], alpha[8];
#pragma unroll
  for (int r = 0; r < 8; ++r) { mrow[r] = -3.0e38f; lrow[r] = 0.f; }

  stage(0, 0);
#ifdef USE_TDM_DEV
  if (w <= 1) __builtin_amdgcn_s_wait_tensorcnt(0);
#endif
  int buf = 0;
  for (int j0 = 0; j0 < jmax; j0 += 32) {
    __syncthreads();                     // tile 'buf' ready, prev readers done
    if (j0 + 32 < jmax) stage(buf ^ 1, j0 + 32);

    if (j0 <= m0w + 15) {                // causal: any useful key in this block
      // scores: S = q(16x64) @ k^T -> two 16x16 tiles
      v8f sc[2];
#pragma unroll
      for (int t = 0; t < 2; ++t) {
        Frag16 kf0, kf1;
        int kr = t * 16 + l16;
        kf0.u4[0] = *(const uint4*)&sK[buf][kr][kb];
        kf0.u4[1] = *(const uint4*)&sK[buf][kr][kb + 16];
        kf1.u4[0] = *(const uint4*)&sK[buf][kr][32 + kb];
        kf1.u4[1] = *(const uint4*)&sK[buf][kr][32 + kb + 16];
        v8f acc = VZ;
        acc = wmma_bf16(qf[0], kf0, acc);
        acc = wmma_bf16(qf[1], kf1, acc);
        sc[t] = acc;
      }
      // key-mask bytes hoisted out of the row loop (index depends on t, lane only)
      bool km0 = msk[b * SEQ + j0 + l16] != 0;
      bool km1 = msk[b * SEQ + j0 + 16 + l16] != 0;
      int  n0j = j0 + l16, n1j = j0 + 16 + l16;
      // mask + running max (shfl_xor butterfly within 16-lane halves)
#pragma unroll
      for (int r = 0; r < 8; ++r) {
        int m = m0w + r + 8 * hlf;
        if (n0j > m || km0) sc[0][r] = -3.0e38f;
        if (n1j > m || km1) sc[1][r] = -3.0e38f;
        float rm = fmaxf(sc[0][r], sc[1][r]);
        rm = fmaxf(rm, __shfl_xor(rm, 1, 32));
        rm = fmaxf(rm, __shfl_xor(rm, 2, 32));
        rm = fmaxf(rm, __shfl_xor(rm, 4, 32));
        rm = fmaxf(rm, __shfl_xor(rm, 8, 32));
        float mn = fmaxf(mrow[r], rm);
        alpha[r] = __expf(mrow[r] - mn);
        mrow[r] = mn;
        sc[0][r] = __expf(sc[0][r] - mn);
        sc[1][r] = __expf(sc[1][r] - mn);
      }
      // P (C layout) -> LDS -> A-layout fragment
#pragma unroll
      for (int r = 0; r < 8; ++r) {
        int rowp = r + 8 * hlf;
        sP[w][rowp][l16]      = (__bf16)sc[0][r];
        sP[w][rowp][16 + l16] = (__bf16)sc[1][r];
      }
      Frag16 pf;
      pf.u4[0] = *(const uint4*)&sP[w][l16][kb];
      pf.u4[1] = *(const uint4*)&sP[w][l16][kb + 16];
      // row sums on the matrix pipe: rs = P @ ones (replicated across half lanes)
      v8f rsv = wmma_bf16(pf, ones, VZ);
#pragma unroll
      for (int r = 0; r < 8; ++r) lrow[r] = lrow[r] * alpha[r] + rsv[r];
      // rescale O by alpha, then accumulate P @ V
#pragma unroll
      for (int nt = 0; nt < 4; ++nt)
#pragma unroll
        for (int r = 0; r < 8; ++r) o[nt][r] = o[nt][r] * alpha[r];
#pragma unroll
      for (int nt = 0; nt < 4; ++nt) {
        Frag16 vf;
        int dr = nt * 16 + l16;
        vf.u4[0] = *(const uint4*)&sVT[buf][dr][kb];
        vf.u4[1] = *(const uint4*)&sVT[buf][dr][kb + 16];
        o[nt] = wmma_bf16(pf, vf, o[nt]);
      }
    }
#ifdef USE_TDM_DEV
    if (w <= 1 && j0 + 32 < jmax) __builtin_amdgcn_s_wait_tensorcnt(0);
#endif
    buf ^= 1;
  }

#pragma unroll
  for (int nt = 0; nt < 4; ++nt)
#pragma unroll
    for (int r = 0; r < 8; ++r) {
      int m = m0w + r + 8 * hlf;
      int d = nt * 16 + l16;
      float val = o[nt][r] / lrow[r];
      O[(size_t)(b * SEQ + m) * (NHEADS * DHEAD) + h * DHEAD + d] = (__bf16)val;
    }
}

// ---------------- host side ----------------
extern "C" void kernel_launch(void* const* d_in, const int* in_sizes, int n_in,
                              void* d_out, int out_size, void* d_ws, size_t ws_size,
                              hipStream_t stream) {
  (void)in_sizes; (void)n_in; (void)out_size; (void)ws_size;
  const float*         x     = (const float*)d_in[0];
  const unsigned char* mask  = (const unsigned char*)d_in[1];   // bool [B,S]
  const float*         gamma = (const float*)d_in[2];
  const float*         Wf    = (const float*)d_in[3];
  const float*         Wao   = (const float*)d_in[4];
  const float*         Wff   = (const float*)d_in[5];
  float*               out   = (float*)d_out;

  char* p = (char*)d_ws;
  __bf16* xn   = (__bf16*)p;  p += (size_t)ROWS * DIMC  * 2;   // 16.8 MB
  __bf16* wfT  = (__bf16*)p;  p += (size_t)FUSED * DIMC * 2;   // 71.8 MB  [FUSED][DIMC]
  __bf16* waoT = (__bf16*)p;  p += (size_t)DIMC * (NHEADS*DHEAD) * 2; // 4.2 MB [2048][1024]
  __bf16* wffT = (__bf16*)p;  p += (size_t)DIMC * FFI   * 2;   // 33.6 MB [2048][8192]
  float*  qkv  = (float*) p;  p += (size_t)ROWS * QKVN  * 4;   // 18.9 MB
  __bf16* qb   = (__bf16*)p;  p += (size_t)ROWS * 1024  * 2;   // 8.4 MB
  __bf16* kbuf = (__bf16*)p;  p += (size_t)ROWS * 64    * 2;   // 0.5 MB
  __bf16* vT   = (__bf16*)p;  p += (size_t)BATCH * 64 * SEQ * 2; // 0.5 MB [b][64][SEQ]
  __bf16* ffb  = (__bf16*)p;  p += (size_t)ROWS * FF2   * 2;   // 134 MB
  __bf16* ao   = (__bf16*)p;  p += (size_t)ROWS * 1024  * 2;   // 8.4 MB

  // weight downcast + transpose (one pass each; B tiles become row-major)
  cast_transpose_kernel<<<dim3(FUSED / 64, DIMC / 64), 256, 0, stream>>>(Wf, wfT, DIMC, FUSED);
  cast_transpose_kernel<<<dim3(DIMC / 64, (NHEADS * DHEAD) / 64), 256, 0, stream>>>(
      Wao, waoT, NHEADS * DHEAD, DIMC);
  cast_transpose_kernel<<<dim3(DIMC / 64, FFI / 64), 256, 0, stream>>>(Wff, wffT, FFI, DIMC);

  // LayerNorm
  ln_kernel<<<ROWS, 256, 0, stream>>>(x, gamma, xn);

  // qkv projection (fp32 out for rope), ff projection (bf16 out)
  gemm_bf16_kernel<0><<<dim3(QKVN / BN, ROWS / BM), 256, 0, stream>>>(
      xn, DIMC, wfT, DIMC, DIMC, qkv, nullptr, QKVN);
  gemm_bf16_kernel<2><<<dim3(FF2 / BN, ROWS / BM), 256, 0, stream>>>(
      xn, DIMC, wfT + (size_t)QKVN * DIMC, DIMC, DIMC, nullptr, ffb, FF2);

  // rope + split (v stored transposed for TDM staging in attention)
  rope_kernel<<<ROWS, 256, 0, stream>>>(qkv, qb, kbuf, vT);

  // attention
  attn_kernel<<<dim3(SEQ / 128, NHEADS, BATCH), 256, 0, stream>>>(qb, kbuf, vT, mask, ao);

  // SwiGLU (in place on first half of ffb)
  glu_kernel<<<(unsigned)(((size_t)ROWS * FFI) / 256), 256, 0, stream>>>(ffb);

  // out = ao @ Wao  (store), then += ffact @ Wff (accumulate)
  gemm_bf16_kernel<0><<<dim3(DIMC / BN, ROWS / BM), 256, 0, stream>>>(
      ao, NHEADS * DHEAD, waoT, NHEADS * DHEAD, NHEADS * DHEAD, out, nullptr, DIMC);
  gemm_bf16_kernel<1><<<dim3(DIMC / BN, ROWS / BM), 256, 0, stream>>>(
      ffb, FF2, wffT, FFI, FFI, out, nullptr, DIMC);
}
